// GPT2LikeLM_48026324303936
// MI455X (gfx1250) — matrix operand order repaired and checked by reading, not compiled, not run on hardware
//
#include <hip/hip_runtime.h>
#include <hip/hip_bf16.h>
#include <cstdint>
#include <cstddef>

#define DEV __device__ __forceinline__

typedef __attribute__((ext_vector_type(16))) __bf16 v16bf;
typedef __attribute__((ext_vector_type(8)))  float  v8f;
typedef int vs4i __attribute__((vector_size(16)));   // matches async-LDS builtin pointee

static constexpr int T_SEQ   = 1024;
static constexpr int DMODEL  = 1024;
static constexpr int DFF     = 4096;
static constexpr int NLAYERS = 6;
static constexpr int VOCAB   = 32000;
static constexpr int MROWS   = 2048;   // B*T

#if defined(__HIP_DEVICE_COMPILE__) && \
    __has_builtin(__builtin_amdgcn_global_load_async_to_lds_b128) && \
    __has_builtin(__builtin_amdgcn_s_wait_asynccnt)
#define USE_ASYNC 1
#else
#define USE_ASYNC 0
#endif

// ---------- helpers ----------
DEV unsigned short f2bf(float f) {            // round-to-nearest-even f32 -> bf16
  unsigned int u = __float_as_uint(f);
  unsigned int r = u + 0x7FFFu + ((u >> 16) & 1u);
  return (unsigned short)(r >> 16);
}

union FragU { uint4 q[2]; v16bf v; };
union U4S8  { uint4 q; unsigned short s[8]; };

DEV v8f wmma_bf16(v16bf a, v16bf b, v8f c) {
  return __builtin_amdgcn_wmma_f32_16x16x32_bf16(false, a, false, b, (short)0, c, false, false);
}

// 16B copy global -> LDS; async (VGPR-free, ASYNCcnt-tracked) when available
#if USE_ASYNC
DEV void cp_b128(const unsigned short* g, unsigned short* l) {
  __builtin_amdgcn_global_load_async_to_lds_b128(
      (__attribute__((address_space(1))) vs4i*)g,
      (__attribute__((address_space(3))) vs4i*)l, 0, 0);
}
DEV void async_wait4() { __builtin_amdgcn_s_wait_asynccnt(4); }
DEV void async_wait0() { __builtin_amdgcn_s_wait_asynccnt(0); }
#else
DEV void cp_b128(const unsigned short* g, unsigned short* l) {
  *(uint4*)l = *(const uint4*)g;
}
DEV void async_wait4() {}
DEV void async_wait0() {}
#endif

// A fragment (16x32 bf16): lane m=lane&15 holds row m; K chunks [8*kg..+7] and [16+8*kg..+7]
DEV v16bf ld_frag_a(const unsigned short* s, int ldr, int row0, int k0, int lane) {
  int m = lane & 15, kg = lane >> 4;
  const unsigned short* base = s + (size_t)(row0 + m) * ldr + k0 + 8 * kg;
  FragU r;
  r.q[0] = *(const uint4*)(base);
  r.q[1] = *(const uint4*)(base + 16);
  return r.v;
}
// B fragment (32x16 bf16) from N-major LDS tile: lane n=lane&15 holds col n; K chunks [8*kg..+7] and [16+8*kg..+7]
DEV v16bf ld_frag_b(const unsigned short* s, int ldr, int n0, int k0, int lane) {
  int n = lane & 15, kg = lane >> 4;
  const unsigned short* base = s + (size_t)(n0 + n) * ldr + k0 + 8 * kg;
  FragU r;
  r.q[0] = *(const uint4*)(base);
  r.q[1] = *(const uint4*)(base + 16);
  return r.v;
}

DEV float hred_max16(float v) {               // reduce over 16-lane half (C-frag row)
  v = fmaxf(v, __shfl_xor(v, 1, 32));
  v = fmaxf(v, __shfl_xor(v, 2, 32));
  v = fmaxf(v, __shfl_xor(v, 4, 32));
  v = fmaxf(v, __shfl_xor(v, 8, 32));
  return v;
}
DEV float hred_sum16(float v) {
  v += __shfl_xor(v, 1, 32);
  v += __shfl_xor(v, 2, 32);
  v += __shfl_xor(v, 4, 32);
  v += __shfl_xor(v, 8, 32);
  return v;
}

// ---------- f32 [K,N] -> bf16 transposed [N,K], batched over blockIdx.z ----------
__global__ __launch_bounds__(256)
void convT_bf16_kernel(const float* __restrict__ src, unsigned short* __restrict__ dst,
                       int K, int N) {
  __shared__ float tile[32][33];
  size_t off = (size_t)blockIdx.z * K * N;
  int k0 = blockIdx.y * 32, n0 = blockIdx.x * 32;
  int tx = threadIdx.x & 31, ty = threadIdx.x >> 5;   // 32 x 8
#pragma unroll
  for (int i = 0; i < 4; ++i)
    tile[ty + 8 * i][tx] = src[off + (size_t)(k0 + ty + 8 * i) * N + n0 + tx];
  __syncthreads();
#pragma unroll
  for (int i = 0; i < 4; ++i)
    dst[off + (size_t)(n0 + ty + 8 * i) * K + k0 + tx] = f2bf(tile[tx][ty + 8 * i]);
}

// ---------- embedding ----------
__global__ __launch_bounds__(256) void embed_kernel(const int* __restrict__ idx,
                                                    const float* __restrict__ tok,
                                                    const float* __restrict__ pos,
                                                    float* __restrict__ x) {
  int row = blockIdx.x;                   // 0..2047
  int t = row & (T_SEQ - 1);
  int tk = idx[row];
  const float* tp = tok + (size_t)tk * DMODEL;
  const float* pp = pos + (size_t)t * DMODEL;
  float* xp = x + (size_t)row * DMODEL;
#pragma unroll
  for (int i = 0; i < 4; ++i) {
    int c = threadIdx.x + 256 * i;
    xp[c] = tp[c] + pp[c];
  }
}

// ---------- layernorm (f32 in, bf16 out) ----------
__global__ __launch_bounds__(256) void layernorm_kernel(const float* __restrict__ x,
                                                        const float* __restrict__ g,
                                                        const float* __restrict__ b,
                                                        unsigned short* __restrict__ out) {
  __shared__ float r1[256], r2[256];
  int row = blockIdx.x, tid = threadIdx.x;
  const float* xp = x + (size_t)row * DMODEL;
  float v[4], s = 0.f, sq = 0.f;
#pragma unroll
  for (int i = 0; i < 4; ++i) {
    v[i] = xp[tid + 256 * i];
    s += v[i]; sq += v[i] * v[i];
  }
  r1[tid] = s; r2[tid] = sq;
  __syncthreads();
  for (int off = 128; off > 0; off >>= 1) {
    if (tid < off) { r1[tid] += r1[tid + off]; r2[tid] += r2[tid + off]; }
    __syncthreads();
  }
  float mu = r1[0] * (1.0f / DMODEL);
  float var = r2[0] * (1.0f / DMODEL) - mu * mu;
  float rstd = rsqrtf(var + 1e-5f);
  unsigned short* op = out + (size_t)row * DMODEL;
#pragma unroll
  for (int i = 0; i < 4; ++i) {
    int c = tid + 256 * i;
    op[c] = f2bf((v[i] - mu) * rstd * g[c] + b[c]);
  }
}

// ---------- WMMA GEMM: C[M,N] = A[M,K](bf16) @ Bt[N,K](bf16)^T (+bias)(+gelu)(+resid) ----------
// A and Bt are both row-contiguous in K, so both tiles stage via async b128 copies.
__global__ __launch_bounds__(256)
void gemm_bf16_kernel(const unsigned short* __restrict__ A,
                      const unsigned short* __restrict__ Bt,
                      const float* __restrict__ bias,
                      const float* __restrict__ resid,
                      float* __restrict__ outF,
                      unsigned short* __restrict__ outB,
                      int M, int N, int K, int act) {
  __shared__ __align__(16) unsigned short As[2][128 * 40];   // [row][k], pad 40
  __shared__ __align__(16) unsigned short Bs[2][128 * 40];   // [n][k],  pad 40
  int tid = threadIdx.x, lane = tid & 31, wave = tid >> 5;
  int wr = wave >> 2, wc = wave & 3;                         // 2 x 4 waves -> 64x32 tiles
  int m0 = blockIdx.y * 128, n0 = blockIdx.x * 128;

  const v8f vzero = {0.f, 0.f, 0.f, 0.f, 0.f, 0.f, 0.f, 0.f};
  v8f acc[4][2];
#pragma unroll
  for (int mi = 0; mi < 4; ++mi)
#pragma unroll
    for (int ni = 0; ni < 2; ++ni) acc[mi][ni] = vzero;

  // stage one K-slab (128x32 A + 128x32 B) into buffer `buf`: 4 async b128 per thread
  auto stage = [&](int buf, int k0) {
#pragma unroll
    for (int i = 0; i < 2; ++i) {
      int q = tid + 256 * i;
      int r = q >> 2, c = q & 3;
      cp_b128(A  + (size_t)(m0 + r) * K + k0 + 8 * c, &As[buf][r * 40 + 8 * c]);
      cp_b128(Bt + (size_t)(n0 + r) * K + k0 + 8 * c, &Bs[buf][r * 40 + 8 * c]);
    }
  };

  int KT = K >> 5;
  stage(0, 0);
  for (int kt = 0; kt < KT; ++kt) {
    int cur = kt & 1;
    bool more = (kt + 1 < KT);
    if (more) { stage(cur ^ 1, (kt + 1) * 32); async_wait4(); }
    else      { async_wait0(); }
    __syncthreads();                         // cur buffer visible to all waves

    v16bf bfr[2];
#pragma unroll
    for (int ni = 0; ni < 2; ++ni) bfr[ni] = ld_frag_b(Bs[cur], 40, wc * 32 + 16 * ni, 0, lane);
#pragma unroll
    for (int mi = 0; mi < 4; ++mi) {
      v16bf afr = ld_frag_a(As[cur], 40, wr * 64 + 16 * mi, 0, lane);
#pragma unroll
      for (int ni = 0; ni < 2; ++ni) acc[mi][ni] = wmma_bf16(afr, bfr[ni], acc[mi][ni]);
    }
    __syncthreads();                         // done reading cur before it is refilled
  }

  int cn = lane & 15, kg = lane >> 4;
#pragma unroll
  for (int mi = 0; mi < 4; ++mi)
#pragma unroll
    for (int ni = 0; ni < 2; ++ni)
#pragma unroll
      for (int j = 0; j < 8; ++j) {
        int gm = m0 + wr * 64 + 16 * mi + j + 8 * kg;
        int gn = n0 + wc * 32 + 16 * ni + cn;
        float v = acc[mi][ni][j];
        if (bias)  v += bias[gn];
        if (act)   v = 0.5f * v * (1.0f + erff(v * 0.70710678118654752f));  // exact GELU
        size_t o = (size_t)gm * N + gn;
        if (resid) v += resid[o];
        if (outF)  outF[o] = v;
        if (outB)  outB[o] = f2bf(v);
      }
}

// ---------- flash attention: one block per (b*H+h, q-block of 128 rows) ----------
__global__ __launch_bounds__(256)
void attn_kernel(const unsigned short* __restrict__ qkv, unsigned short* __restrict__ y) {
  __shared__ __align__(16) unsigned short Qs[128 * 72];    // [qrow][d]
  __shared__ __align__(16) unsigned short Ks[64 * 72];     // [kv][d]  (B-transposed for Q.K^T)
  __shared__ __align__(16) unsigned short Vt[64 * 72];     // [d][kv]  (B-transposed for P.V)
  __shared__ __align__(16) unsigned short Ps[8 * 16 * 72]; // per-wave P strip [16][64+pad]

  int tid = threadIdx.x, lane = tid & 31, wave = tid >> 5;
  int cn = lane & 15, kg = lane >> 4;
  int bh = blockIdx.x;                 // 0..31
  int b = bh >> 4, h = bh & 15;
  int q0 = blockIdx.y * 128;
  int bT = b * T_SEQ, h64 = h * 64;

  // stage Q tile 128x64 (async b128, 4 per thread)
  for (int i = tid; i < 1024; i += 256) {
    int r = i >> 3, c = i & 7;
    cp_b128(qkv + (size_t)(bT + q0 + r) * (3 * DMODEL) + h64 + 8 * c, &Qs[r * 72 + 8 * c]);
  }
  async_wait0();
  __syncthreads();

  const v8f vzero = {0.f, 0.f, 0.f, 0.f, 0.f, 0.f, 0.f, 0.f};
  float mrow[8], lrow[8];
  v8f o[4];
#pragma unroll
  for (int j = 0; j < 8; ++j) { mrow[j] = -3.0e38f; lrow[j] = 0.f; }
#pragma unroll
  for (int dt = 0; dt < 4; ++dt) o[dt] = vzero;

  int kbmax = (q0 + 127) >> 6;          // inclusive: kv blocks of 64
  for (int kb = 0; kb <= kbmax; ++kb) {
    int kv0 = kb * 64;
    // stage K tile row-major [kv][d] (async b128, 2/thread)
    for (int i = tid; i < 512; i += 256) {
      int r = i >> 3, c = i & 7;
      cp_b128(qkv + (size_t)(bT + kv0 + r) * (3 * DMODEL) + DMODEL + h64 + 8 * c,
              &Ks[r * 72 + 8 * c]);
    }
    // stage V transposed [d][kv] (small scalar scatter)
    {
      int kv = tid >> 2, dc = tid & 3;
      const unsigned short* gp =
          qkv + (size_t)(bT + kv0 + kv) * (3 * DMODEL) + 2 * DMODEL + h64 + 16 * dc;
      U4S8 a0, a1;
      a0.q = *(const uint4*)gp;
      a1.q = *(const uint4*)(gp + 8);
#pragma unroll
      for (int e = 0; e < 8; ++e) {
        Vt[(16 * dc + e) * 72 + kv]     = a0.s[e];
        Vt[(16 * dc + 8 + e) * 72 + kv] = a1.s[e];
      }
    }
    async_wait0();
    __syncthreads();

    // S strip (16 q-rows x 64 kv) = Q_strip @ K^T
    v8f sA[4];
#pragma unroll
    for (int nt = 0; nt < 4; ++nt) sA[nt] = vzero;
#pragma unroll
    for (int ks = 0; ks < 2; ++ks) {
      v16bf aq = ld_frag_a(Qs, 72, 16 * wave, 32 * ks, lane);
#pragma unroll
      for (int nt = 0; nt < 4; ++nt) {
        v16bf bk = ld_frag_b(Ks, 72, 16 * nt, 32 * ks, lane);
        sA[nt] = wmma_bf16(aq, bk, sA[nt]);
      }
    }
    // scale + causal mask
    bool need_mask = (kv0 + 63 > q0);
#pragma unroll
    for (int nt = 0; nt < 4; ++nt)
#pragma unroll
      for (int j = 0; j < 8; ++j) {
        float v = sA[nt][j] * 0.125f;    // 1/sqrt(64)
        if (need_mask) {
          int qrow = q0 + 16 * wave + j + 8 * kg;
          int kcol = kv0 + 16 * nt + cn;
          if (kcol > qrow) v = -3.0e38f;
        }
        sA[nt][j] = v;
      }
    // online softmax update (row = C-frag VGPR j + 8*kg, reduce over 16-lane half)
#pragma unroll
    for (int j = 0; j < 8; ++j) {
      float mx = fmaxf(fmaxf(sA[0][j], sA[1][j]), fmaxf(sA[2][j], sA[3][j]));
      mx = hred_max16(mx);
      float mn = fmaxf(mrow[j], mx);
      float alpha = __expf(mrow[j] - mn);
      mrow[j] = mn;
      float rs = 0.f;
#pragma unroll
      for (int nt = 0; nt < 4; ++nt) {
        float p = __expf(sA[nt][j] - mn);
        sA[nt][j] = p;
        rs += p;
      }
      rs = hred_sum16(rs);
      lrow[j] = lrow[j] * alpha + rs;
#pragma unroll
      for (int dt = 0; dt < 4; ++dt) o[dt][j] *= alpha;
    }
    // restage P as bf16 A-operand
    unsigned short* pw = &Ps[wave * 16 * 72];
#pragma unroll
    for (int nt = 0; nt < 4; ++nt)
#pragma unroll
      for (int j = 0; j < 8; ++j)
        pw[(j + 8 * kg) * 72 + 16 * nt + cn] = f2bf(sA[nt][j]);
    __syncthreads();
    // O += P @ V
#pragma unroll
    for (int ks = 0; ks < 2; ++ks) {
      v16bf ap = ld_frag_a(pw, 72, 0, 32 * ks, lane);
#pragma unroll
      for (int dt = 0; dt < 4; ++dt) {
        v16bf bv = ld_frag_b(Vt, 72, 16 * dt, 32 * ks, lane);
        o[dt] = wmma_bf16(ap, bv, o[dt]);
      }
    }
    __syncthreads();
  }

  // normalize + store y[b,t, h*64+d] as bf16
#pragma unroll
  for (int j = 0; j < 8; ++j) {
    float inv = 1.0f / lrow[j];
    int t = q0 + 16 * wave + j + 8 * kg;
#pragma unroll
    for (int dt = 0; dt < 4; ++dt) {
      int d = 16 * dt + cn;
      y[(size_t)(bT + t) * DMODEL + h64 + d] = f2bf(o[dt][j] * inv);
    }
  }
}

// ---------- host launcher ----------
extern "C" void kernel_launch(void* const* d_in, const int* in_sizes, int n_in,
                              void* d_out, int out_size, void* d_ws, size_t ws_size,
                              hipStream_t stream) {
  (void)in_sizes; (void)n_in; (void)out_size; (void)ws_size;
  const int*   idx     = (const int*)d_in[0];
  const float* tok_emb = (const float*)d_in[1];
  const float* pos_emb = (const float*)d_in[2];
  const float* ln1_s   = (const float*)d_in[3];
  const float* ln1_b   = (const float*)d_in[4];
  const float* qkv_w   = (const float*)d_in[5];
  const float* qkv_b   = (const float*)d_in[6];
  const float* out_w   = (const float*)d_in[7];
  const float* out_b   = (const float*)d_in[8];
  const float* ln2_s   = (const float*)d_in[9];
  const float* ln2_b   = (const float*)d_in[10];
  const float* mlp_w1  = (const float*)d_in[11];
  const float* mlp_b1  = (const float*)d_in[12];
  const float* mlp_w2  = (const float*)d_in[13];
  const float* mlp_b2  = (const float*)d_in[14];
  const float* lnf_s   = (const float*)d_in[15];
  const float* lnf_b   = (const float*)d_in[16];
  const float* head_w  = (const float*)d_in[17];

  uintptr_t base = (uintptr_t)d_ws;
  auto take = [&](size_t bytes) -> void* {
    uintptr_t p = base;
    base += (bytes + 255) & ~(size_t)255;
    return (void*)p;
  };
  const size_t nQKV = (size_t)NLAYERS * DMODEL * 3 * DMODEL;
  const size_t nOUT = (size_t)NLAYERS * DMODEL * DMODEL;
  const size_t nW1  = (size_t)NLAYERS * DMODEL * DFF;
  const size_t nW2  = (size_t)NLAYERS * DFF * DMODEL;
  const size_t nWH  = (size_t)DMODEL * VOCAB;

  unsigned short* wq = (unsigned short*)take(nQKV * 2);   // all weights stored [N][K] bf16
  unsigned short* wo = (unsigned short*)take(nOUT * 2);
  unsigned short* w1 = (unsigned short*)take(nW1 * 2);
  unsigned short* w2 = (unsigned short*)take(nW2 * 2);
  unsigned short* wh = (unsigned short*)take(nWH * 2);
  float*          x  = (float*)take((size_t)MROWS * DMODEL * 4);
  unsigned short* hb = (unsigned short*)take((size_t)MROWS * DMODEL * 2);
  unsigned short* qk = (unsigned short*)take((size_t)MROWS * 3 * DMODEL * 2);
  unsigned short* yb = (unsigned short*)take((size_t)MROWS * DMODEL * 2);
  unsigned short* mb = (unsigned short*)take((size_t)MROWS * DFF * 2);

  // weights f32 [K,N] -> bf16 transposed [N,K]
  convT_bf16_kernel<<<dim3(3 * DMODEL / 32, DMODEL / 32, NLAYERS), 256, 0, stream>>>(
      qkv_w, wq, DMODEL, 3 * DMODEL);
  convT_bf16_kernel<<<dim3(DMODEL / 32, DMODEL / 32, NLAYERS), 256, 0, stream>>>(
      out_w, wo, DMODEL, DMODEL);
  convT_bf16_kernel<<<dim3(DFF / 32, DMODEL / 32, NLAYERS), 256, 0, stream>>>(
      mlp_w1, w1, DMODEL, DFF);
  convT_bf16_kernel<<<dim3(DMODEL / 32, DFF / 32, NLAYERS), 256, 0, stream>>>(
      mlp_w2, w2, DFF, DMODEL);
  convT_bf16_kernel<<<dim3(VOCAB / 32, DMODEL / 32, 1), 256, 0, stream>>>(
      head_w, wh, DMODEL, VOCAB);

  embed_kernel<<<MROWS, 256, 0, stream>>>(idx, tok_emb, pos_emb, x);

  for (int l = 0; l < NLAYERS; ++l) {
    layernorm_kernel<<<MROWS, 256, 0, stream>>>(x, ln1_s + l * DMODEL, ln1_b + l * DMODEL, hb);
    gemm_bf16_kernel<<<dim3(3 * DMODEL / 128, MROWS / 128), 256, 0, stream>>>(
        hb, wq + (size_t)l * DMODEL * 3 * DMODEL, qkv_b + (size_t)l * 3 * DMODEL,
        nullptr, nullptr, qk, MROWS, 3 * DMODEL, DMODEL, 0);
    attn_kernel<<<dim3(32, T_SEQ / 128), 256, 0, stream>>>(qk, yb);
    gemm_bf16_kernel<<<dim3(DMODEL / 128, MROWS / 128), 256, 0, stream>>>(
        yb, wo + (size_t)l * DMODEL * DMODEL, out_b + (size_t)l * DMODEL,
        x, x, nullptr, MROWS, DMODEL, DMODEL, 0);
    layernorm_kernel<<<MROWS, 256, 0, stream>>>(x, ln2_s + l * DMODEL, ln2_b + l * DMODEL, hb);
    gemm_bf16_kernel<<<dim3(DFF / 128, MROWS / 128), 256, 0, stream>>>(
        hb, w1 + (size_t)l * DMODEL * DFF, mlp_b1 + (size_t)l * DFF,
        nullptr, nullptr, mb, MROWS, DFF, DMODEL, 1);
    gemm_bf16_kernel<<<dim3(DMODEL / 128, MROWS / 128), 256, 0, stream>>>(
        mb, w2 + (size_t)l * DFF * DMODEL, mlp_b2 + (size_t)l * DMODEL,
        x, x, nullptr, MROWS, DMODEL, DFF, 0);
  }
  layernorm_kernel<<<MROWS, 256, 0, stream>>>(x, lnf_s, lnf_b, hb);
  gemm_bf16_kernel<<<dim3(VOCAB / 128, MROWS / 128), 256, 0, stream>>>(
      hb, wh, nullptr, nullptr, (float*)d_out, nullptr, MROWS, VOCAB, DMODEL, 0);
}
